// ErwinTransformerBlock_55611236549029
// MI455X (gfx1250) — compile-verified, matrix operand order checked by reading, not run on hardware
//
#include <hip/hip_runtime.h>
#include <hip/hip_bf16.h>

// ---------------------------------------------------------------------------
// Erwin transformer block on gfx1250 (MI455X).
// bf16 WMMA pipeline + TDM (tensor_load_to_lds) staging for attention tiles.
// ---------------------------------------------------------------------------

#define DIM      256
#define HEADS    8
#define BALL     128
#define HEAD_DIM 32
#define HIDDEN   1024
#define EPSV     1e-6f

typedef __bf16 bf16_t;
typedef __attribute__((ext_vector_type(16))) __bf16 v16bf;
typedef __attribute__((ext_vector_type(8)))  float  v8f;
typedef __attribute__((ext_vector_type(4)))  unsigned int v4u;
typedef __attribute__((ext_vector_type(8)))  int v8i;
typedef __attribute__((ext_vector_type(4)))  int v4i;

__device__ __forceinline__ v8f zero_v8f() {
  v8f z;
#pragma unroll
  for (int i = 0; i < 8; ++i) z[i] = 0.0f;
  return z;
}

// WMMA wrapper: (neg_a, A, neg_b, B, c_mod, C, reuse_a, reuse_b)
__device__ __forceinline__ v8f wmma_bf16(v16bf a, v16bf b, v8f c) {
  return __builtin_amdgcn_wmma_f32_16x16x32_bf16(false, a, false, b,
                                                 (short)0, c, false, false);
}

// A-style fragment (also B for "result = X @ W^T": B[k][n] = W[n][k]).
// ISA 7.12.2 (16-bit A 16x32, wave32):
//   lane L: row = row0 + (L&15); element e -> k = k0 + (e>>3)*16 + (L>>4)*8 + (e&7)
__device__ __forceinline__ v16bf frag_rm(const bf16_t* src, int ld, int row0, int k0) {
  const int lane = (int)(threadIdx.x & 31u);
  const bf16_t* p = src + (long)(row0 + (lane & 15)) * ld + k0 + ((lane >> 4) << 3);
  v16bf f;
#pragma unroll
  for (int i = 0; i < 8; ++i) { f[i] = p[i]; f[i + 8] = p[i + 16]; }
  return f;
}

__device__ __forceinline__ int acc_row_off(int r) {
  return r + ((((int)threadIdx.x & 31) >> 4) << 3);
}
__device__ __forceinline__ int acc_col_off() { return (int)threadIdx.x & 15; }

// ---------------------------------------------------------------------------
// TDM: 1-D contiguous DMA global->LDS, ndw dwords (data_size=4B).
// Descriptor per CDNA5 ISA (D# groups 0/1; groups 2/3 zero).
// ---------------------------------------------------------------------------
#if __has_builtin(__builtin_amdgcn_tensor_load_to_lds)
#define USE_TDM 1
__device__ __forceinline__ void tdm_copy_1d(unsigned lds_off, const void* gsrc,
                                            unsigned ndw) {
  const unsigned long ga = (unsigned long)gsrc;
  v4u g0;
  g0[0] = 1u;                                        // count=1, is_restore=0
  g0[1] = lds_off;                                   // lds_addr (bytes)
  g0[2] = (unsigned)ga;                              // global_addr[31:0]
  g0[3] = (unsigned)((ga >> 32) & 0x01FFFFFFu) | (2u << 30); // addr[56:32] | type=2
  v8i g1;
  g1[0] = (int)(2u << 16);                           // data_size = 4 bytes
  g1[1] = (int)((ndw & 0xFFFFu) << 16);              // tensor_dim0[15:0]
  g1[2] = (int)((ndw >> 16) & 0xFFFFu) | (1 << 16);  // tensor_dim0[31:16] | tensor_dim1=1
  g1[3] = (int)((ndw & 0xFFFFu) << 16);              // tile_dim0
  g1[4] = 1;                                         // tile_dim1 = 1
  g1[5] = (int)ndw;                                  // tensor_dim0_stride[31:0]
  g1[6] = 0;
  g1[7] = 0;
  v4i z4 = {0, 0, 0, 0};
#if __clang_major__ >= 23
  v8i z8 = {0, 0, 0, 0, 0, 0, 0, 0};
  __builtin_amdgcn_tensor_load_to_lds(g0, g1, z4, z4, z8, 0);
#else
  __builtin_amdgcn_tensor_load_to_lds(g0, g1, z4, z4, 0);
#endif
}
__device__ __forceinline__ void tdm_wait0() {
#if __has_builtin(__builtin_amdgcn_s_wait_tensorcnt)
  __builtin_amdgcn_s_wait_tensorcnt(0);
#else
  asm volatile("s_wait_tensorcnt 0x0" ::: "memory");
#endif
}
#else
#define USE_TDM 0
#endif

// ---------------------------------------------------------------------------
// Kernel 1: fp32 -> bf16 weight conversion
// ---------------------------------------------------------------------------
__global__ void cvt_f32_bf16(const float* __restrict__ src, bf16_t* __restrict__ dst, int n) {
  int i = blockIdx.x * blockDim.x + threadIdx.x;
  if (i < n) dst[i] = (bf16_t)src[i];
}

// ---------------------------------------------------------------------------
// Kernel 2: ball pos-mean + RMSNorm(norm1) + pos-encode + QKV GEMM (256->768)
// Writes head-major Q/K and per-(head,ball) transposed V so attention tiles
// are contiguous 8KB blocks (TDM-friendly).
// dyn LDS: xn[128*256] bf16 | partial[256] f32 | mean[4] f32
// ---------------------------------------------------------------------------
__global__ __launch_bounds__(256) void norm_pe_qkv_kernel(
    const float* __restrict__ x, const float* __restrict__ pos,
    const float* __restrict__ norm1_w, const float* __restrict__ pe_w,
    const float* __restrict__ pe_b, const float* __restrict__ qkv_b,
    const bf16_t* __restrict__ qkv_w_bf,
    bf16_t* __restrict__ Qh, bf16_t* __restrict__ Kh, bf16_t* __restrict__ Vth,
    long N, int nballs)
{
  extern __shared__ char smem[];
  bf16_t* xn    = (bf16_t*)smem;                    // 65536 B
  float*  part  = (float*)(smem + BALL * DIM * 2);  // 1024 B
  float*  meanp = part + 256;                       // 16 B

  const int  tid  = (int)threadIdx.x;
  const int  ball = (int)blockIdx.x;
  const long base = (long)ball * BALL;

  if (tid < 3) {
    float s = 0.0f;
    for (int i = 0; i < BALL; ++i) s += pos[(base + i) * 3 + tid];
    meanp[tid] = s * (1.0f / BALL);
  }
  __syncthreads();
  const float m0 = meanp[0], m1 = meanp[1], m2 = meanp[2];

  const int row = tid >> 1, half = tid & 1;
  const float* xr = x + (base + row) * DIM;
  float ss = 0.0f;
  for (int c = half * 128; c < half * 128 + 128; ++c) { float v = xr[c]; ss += v * v; }
  part[tid] = ss;
  __syncthreads();
  const float inv = rsqrtf((part[row * 2] + part[row * 2 + 1]) * (1.0f / DIM) + EPSV);
  const float rx = pos[(base + row) * 3 + 0] - m0;
  const float ry = pos[(base + row) * 3 + 1] - m1;
  const float rz = pos[(base + row) * 3 + 2] - m2;
  for (int c = half * 128; c < half * 128 + 128; ++c) {
    float pe = pe_w[c * 3 + 0] * rx + pe_w[c * 3 + 1] * ry + pe_w[c * 3 + 2] * rz + pe_b[c];
    xn[row * DIM + c] = (bf16_t)(xr[c] * inv * norm1_w[c] + pe);
  }
  __syncthreads();

  // GEMM: xn(128x256) @ qkv_w^T(256x768); qkv col = h*96 + e*3 + {q,k,v}
  const int wv = tid >> 5;
  __builtin_prefetch(qkv_w_bf + (long)(wv * 6 * 16) * DIM, 0, 3);
  for (int mt = 0; mt < 8; ++mt) {
    v16bf a[8];
#pragma unroll
    for (int k = 0; k < 8; ++k) a[k] = frag_rm(xn, DIM, mt * 16, k * 32);
    for (int j = 0; j < 6; ++j) {
      const int n0 = (wv * 6 + j) * 16;
      v8f acc = zero_v8f();
#pragma unroll
      for (int k = 0; k < 8; ++k)
        acc = wmma_bf16(a[k], frag_rm(qkv_w_bf, DIM, n0, k * 32), acc);
      const int col  = n0 + acc_col_off();
      const float bias = qkv_b[col];
      const int h = col / 96, rem = col % 96;
      const int e = rem / 3, which = rem % 3;
#pragma unroll
      for (int r = 0; r < 8; ++r) {
        const int lr = mt * 16 + acc_row_off(r);
        const bf16_t v = (bf16_t)(acc[r] + bias);
        if (which == 0)
          Qh[((long)h * N + base + lr) * HEAD_DIM + e] = v;
        else if (which == 1)
          Kh[((long)h * N + base + lr) * HEAD_DIM + e] = v;
        else
          Vth[(((long)h * nballs + ball) * HEAD_DIM + e) * BALL + lr] = v;
      }
    }
  }
}

// ---------------------------------------------------------------------------
// Kernel 3: per-(ball, head) attention. grid = nballs*8, 256 threads = 8 waves.
// Q/K/Vt tiles are contiguous 8KB blocks -> staged with 3 TDM descriptors.
// ---------------------------------------------------------------------------
__global__ __launch_bounds__(256) void ball_attn_kernel(
    const bf16_t* __restrict__ Qh, const bf16_t* __restrict__ Kh,
    const bf16_t* __restrict__ Vth, const float* __restrict__ pos,
    const float* __restrict__ sigma_att, bf16_t* __restrict__ attn_out,
    long N, int nballs)
{
  // single LDS blob so TDM lds_addr offsets are well-defined
  __shared__ char lds3[8192 * 3 + 1536 + 8 * 16 * BALL * 2];
  bf16_t* Qs   = (bf16_t*)(lds3);             // 8 KB
  bf16_t* Ks   = (bf16_t*)(lds3 + 8192);      // 8 KB
  bf16_t* Vt   = (bf16_t*)(lds3 + 16384);     // 8 KB  (Vt[e][tok])
  float*  posL = (float*)(lds3 + 24576);      // 1.5 KB
  bf16_t* Ps   = (bf16_t*)(lds3 + 26112);     // 32 KB (wave-private P rows)

  const int  ball = (int)blockIdx.x >> 3;
  const int  head = (int)blockIdx.x & 7;
  const int  tid  = (int)threadIdx.x;
  const long base = (long)ball * BALL;

  const bf16_t* gQ = Qh + ((long)head * N + base) * HEAD_DIM;
  const bf16_t* gK = Kh + ((long)head * N + base) * HEAD_DIM;
  const bf16_t* gV = Vth + ((long)head * nballs + ball) * (HEAD_DIM * BALL);

#if USE_TDM
  if (tid < 32) {                 // wave 0 issues the three DMAs
    tdm_copy_1d(0u,     gQ, 2048u);
    tdm_copy_1d(8192u,  gK, 2048u);
    tdm_copy_1d(16384u, gV, 2048u);
    tdm_wait0();
  }
#else
  for (int c = tid; c < 512; c += 256) {
    ((uint4*)Qs)[c] = ((const uint4*)gQ)[c];
    ((uint4*)Ks)[c] = ((const uint4*)gK)[c];
    ((uint4*)Vt)[c] = ((const uint4*)gV)[c];
  }
#endif
  for (int i = tid; i < BALL * 3; i += 256) posL[i] = pos[base * 3 + i];
  __syncthreads();

  const int   wv    = tid >> 5;
  const int   lane  = tid & 31;
  const int   m0    = wv * 16;
  const int   hi    = (lane >> 4) << 3;
  const float sigma = sigma_att[head];
  const float scale = 0.17677669529663687f; // 1/sqrt(32)

  const v16bf aQ = frag_rm(Qs, HEAD_DIM, m0, 0);
  float s[8][8];
#pragma unroll
  for (int t = 0; t < 8; ++t) {
    const v16bf bK = frag_rm(Ks, HEAD_DIM, t * 16, 0);
    const v8f acc = wmma_bf16(aQ, bK, zero_v8f());
    const int nn = t * 16 + (lane & 15);
    const float px = posL[nn * 3], py = posL[nn * 3 + 1], pz = posL[nn * 3 + 2];
#pragma unroll
    for (int r = 0; r < 8; ++r) {
      const int mm = m0 + r + hi;
      const float dx = posL[mm * 3] - px, dy = posL[mm * 3 + 1] - py, dz = posL[mm * 3 + 2] - pz;
      s[t][r] = acc[r] * scale + sigma * sqrtf(dx * dx + dy * dy + dz * dz);
    }
  }

  // Row softmax: shfl masks 8..1 stay inside the 16-lane halves (wave32)
  float rinv[8];
#pragma unroll
  for (int r = 0; r < 8; ++r) {
    float mx = -1e30f;
#pragma unroll
    for (int t = 0; t < 8; ++t) mx = fmaxf(mx, s[t][r]);
    for (int off = 8; off >= 1; off >>= 1) mx = fmaxf(mx, __shfl_xor(mx, off, 32));
    float sum = 0.0f;
#pragma unroll
    for (int t = 0; t < 8; ++t) { const float e = __expf(s[t][r] - mx); s[t][r] = e; sum += e; }
    for (int off = 8; off >= 1; off >>= 1) sum += __shfl_xor(sum, off, 32);
    rinv[r] = 1.0f / sum;
  }

  // Stage P (bf16) into wave-private LDS, then out = P(16x128) @ V(128x32)
  bf16_t* Pw = Ps + wv * 16 * BALL;
#pragma unroll
  for (int t = 0; t < 8; ++t)
#pragma unroll
    for (int r = 0; r < 8; ++r)
      Pw[(r + hi) * BALL + t * 16 + (lane & 15)] = (bf16_t)(s[t][r] * rinv[r]);

#pragma unroll
  for (int nt = 0; nt < 2; ++nt) {
    v8f acc = zero_v8f();
#pragma unroll
    for (int k = 0; k < 4; ++k)
      acc = wmma_bf16(frag_rm(Pw, BALL, 0, k * 32),
                      frag_rm(Vt, BALL, nt * 16, k * 32), acc);
    const int col = head * 32 + nt * 16 + (lane & 15);
#pragma unroll
    for (int r = 0; r < 8; ++r)
      attn_out[(base + m0 + r + hi) * DIM + col] = (bf16_t)acc[r];
  }
}

// ---------------------------------------------------------------------------
// Kernel 4: proj GEMM (256x256) + bias + residual -> x1 (fp32)
// ---------------------------------------------------------------------------
__global__ __launch_bounds__(256) void proj_res_kernel(
    const bf16_t* __restrict__ attn, const bf16_t* __restrict__ proj_w_bf,
    const float* __restrict__ proj_b, const float* __restrict__ x,
    float* __restrict__ x1)
{
  const long base = (long)blockIdx.x * BALL;
  const int  tid  = (int)threadIdx.x, wv = tid >> 5, lane = tid & 31;
  const bf16_t* A = attn + base * DIM;
  __builtin_prefetch(proj_w_bf, 0, 3);
  for (int mt = 0; mt < 8; ++mt) {
    v16bf a[8];
#pragma unroll
    for (int k = 0; k < 8; ++k) a[k] = frag_rm(A, DIM, mt * 16, k * 32);
#pragma unroll
    for (int j = 0; j < 2; ++j) {
      const int n0 = (wv * 2 + j) * 16;
      v8f acc = zero_v8f();
#pragma unroll
      for (int k = 0; k < 8; ++k)
        acc = wmma_bf16(a[k], frag_rm(proj_w_bf, DIM, n0, k * 32), acc);
      const int col = n0 + (lane & 15);
      const float bias = proj_b[col];
#pragma unroll
      for (int r = 0; r < 8; ++r) {
        const long tok = base + mt * 16 + r + ((lane >> 4) << 3);
        x1[tok * DIM + col] = acc[r] + bias + x[tok * DIM + col];
      }
    }
  }
}

// ---------------------------------------------------------------------------
// Kernel 5: RMSNorm(norm2) + w1/w2 GEMMs (256->1024) + SiLU*val -> g (bf16)
// dyn LDS: h[128*256] bf16 | partial[256] f32
// ---------------------------------------------------------------------------
__global__ __launch_bounds__(256) void mlp_gate_kernel(
    const float* __restrict__ x1, const float* __restrict__ norm2_w,
    const bf16_t* __restrict__ w1_bf, const float* __restrict__ w1_b,
    const bf16_t* __restrict__ w2_bf, const float* __restrict__ w2_b,
    bf16_t* __restrict__ g)
{
  extern __shared__ char smem[];
  bf16_t* h    = (bf16_t*)smem;
  float*  part = (float*)(smem + BALL * DIM * 2);

  const long base = (long)blockIdx.x * BALL;
  const int  tid  = (int)threadIdx.x;
  const int  row  = tid >> 1, half = tid & 1;
  const float* xr = x1 + (base + row) * DIM;
  float ss = 0.0f;
  for (int c = half * 128; c < half * 128 + 128; ++c) { float v = xr[c]; ss += v * v; }
  part[tid] = ss;
  __syncthreads();
  const float inv = rsqrtf((part[row * 2] + part[row * 2 + 1]) * (1.0f / DIM) + EPSV);
  for (int c = half * 128; c < half * 128 + 128; ++c)
    h[row * DIM + c] = (bf16_t)(xr[c] * inv * norm2_w[c]);
  __syncthreads();

  const int wv = tid >> 5, lane = tid & 31;
  bf16_t* gout = g + base * HIDDEN;
  __builtin_prefetch(w1_bf + (long)(wv * 8 * 16) * DIM, 0, 3);
  for (int mt = 0; mt < 8; ++mt) {
    v16bf a[8];
#pragma unroll
    for (int k = 0; k < 8; ++k) a[k] = frag_rm(h, DIM, mt * 16, k * 32);
    for (int j = 0; j < 8; ++j) {
      const int n0 = (wv * 8 + j) * 16;
      v8f acc1 = zero_v8f(), acc2 = zero_v8f();
#pragma unroll
      for (int k = 0; k < 8; ++k) {
        acc1 = wmma_bf16(a[k], frag_rm(w1_bf, DIM, n0, k * 32), acc1);
        acc2 = wmma_bf16(a[k], frag_rm(w2_bf, DIM, n0, k * 32), acc2);
      }
      const int col = n0 + (lane & 15);
      const float b1 = w1_b[col], b2 = w2_b[col];
#pragma unroll
      for (int r = 0; r < 8; ++r) {
        const float gate = acc1[r] + b1;
        const float val  = acc2[r] + b2;
        const float sig  = 1.0f / (1.0f + __expf(-gate));
        const int   lr   = mt * 16 + r + ((lane >> 4) << 3);
        gout[(long)lr * HIDDEN + col] = (bf16_t)(gate * sig * val);
      }
    }
  }
}

// ---------------------------------------------------------------------------
// Kernel 6: w3 GEMM (1024->256) + bias + residual -> out (fp32)
// ---------------------------------------------------------------------------
__global__ __launch_bounds__(256) void mlp_out_kernel(
    const bf16_t* __restrict__ g, const bf16_t* __restrict__ w3_bf,
    const float* __restrict__ w3_b, const float* __restrict__ x1,
    float* __restrict__ out)
{
  const long base = (long)blockIdx.x * BALL;
  const int  tid  = (int)threadIdx.x, wv = tid >> 5, lane = tid & 31;
  const bf16_t* A = g + base * HIDDEN;
  __builtin_prefetch(w3_bf + (long)(wv * 2 * 16) * HIDDEN, 0, 3);
  for (int mt = 0; mt < 8; ++mt) {
    v8f acc[2]; acc[0] = zero_v8f(); acc[1] = zero_v8f();
    for (int kc = 0; kc < 4; ++kc) {          // K = 1024 in 4 chunks of 256
      v16bf a[8];
#pragma unroll
      for (int k = 0; k < 8; ++k) a[k] = frag_rm(A, HIDDEN, mt * 16, kc * 256 + k * 32);
#pragma unroll
      for (int j = 0; j < 2; ++j) {
        const int n0 = (wv * 2 + j) * 16;
#pragma unroll
        for (int k = 0; k < 8; ++k)
          acc[j] = wmma_bf16(a[k], frag_rm(w3_bf, HIDDEN, n0, kc * 256 + k * 32), acc[j]);
      }
    }
#pragma unroll
    for (int j = 0; j < 2; ++j) {
      const int col = (wv * 2 + j) * 16 + (lane & 15);
      const float bias = w3_b[col];
#pragma unroll
      for (int r = 0; r < 8; ++r) {
        const long tok = base + mt * 16 + r + ((lane >> 4) << 3);
        out[tok * DIM + col] = acc[j][r] + bias + x1[tok * DIM + col];
      }
    }
  }
}

// ---------------------------------------------------------------------------
// Host launcher
// ---------------------------------------------------------------------------
extern "C" void kernel_launch(void* const* d_in, const int* in_sizes, int n_in,
                              void* d_out, int out_size, void* d_ws, size_t ws_size,
                              hipStream_t stream)
{
  const float* x       = (const float*)d_in[0];
  const float* pos     = (const float*)d_in[1];
  // d_in[2] = num_batches (unused by the math)
  const float* qkv_w   = (const float*)d_in[3];
  const float* qkv_b   = (const float*)d_in[4];
  const float* proj_w  = (const float*)d_in[5];
  const float* proj_b  = (const float*)d_in[6];
  const float* pe_w    = (const float*)d_in[7];
  const float* pe_b    = (const float*)d_in[8];
  const float* sigma   = (const float*)d_in[9];
  const float* norm1_w = (const float*)d_in[10];
  const float* norm2_w = (const float*)d_in[11];
  const float* w1_w    = (const float*)d_in[12];
  const float* w1_b    = (const float*)d_in[13];
  const float* w2_w    = (const float*)d_in[14];
  const float* w2_b    = (const float*)d_in[15];
  const float* w3_w    = (const float*)d_in[16];
  const float* w3_b    = (const float*)d_in[17];

  const long N      = (long)in_sizes[0] / DIM;
  const int  nballs = (int)(N / BALL);

  // Workspace layout:
  char* ws = (char*)d_ws;
  bf16_t* qkv_w_bf  = (bf16_t*)ws; ws += sizeof(bf16_t) * 3 * DIM * DIM;   // 768x256
  bf16_t* proj_w_bf = (bf16_t*)ws; ws += sizeof(bf16_t) * DIM * DIM;       // 256x256
  bf16_t* w1_bf     = (bf16_t*)ws; ws += sizeof(bf16_t) * HIDDEN * DIM;    // 1024x256
  bf16_t* w2_bf     = (bf16_t*)ws; ws += sizeof(bf16_t) * HIDDEN * DIM;    // 1024x256
  bf16_t* w3_bf     = (bf16_t*)ws; ws += sizeof(bf16_t) * DIM * HIDDEN;    // 256x1024
  float*  x1        = (float*)ws;  ws += sizeof(float) * N * DIM;          // 128 MB
  bf16_t* Qh        = (bf16_t*)ws;                                         // 64 MB
  bf16_t* Kh        = Qh + (long)N * HEAD_DIM * HEADS;                     // 64 MB
  bf16_t* Vth       = Kh + (long)N * HEAD_DIM * HEADS;                     // 64 MB
  bf16_t* attn_act  = Vth + (long)N * HEAD_DIM * HEADS;                    // 64 MB
  bf16_t* g_act     = Qh;     // reuse Q/K/V/attn span (256 MB) after kernel 4

  // 1) weight conversion
  {
    const int T = 256;
    cvt_f32_bf16<<<(3 * DIM * DIM + T - 1) / T, T, 0, stream>>>(qkv_w,  qkv_w_bf,  3 * DIM * DIM);
    cvt_f32_bf16<<<(DIM * DIM + T - 1) / T,     T, 0, stream>>>(proj_w, proj_w_bf, DIM * DIM);
    cvt_f32_bf16<<<(HIDDEN * DIM + T - 1) / T,  T, 0, stream>>>(w1_w,   w1_bf,     HIDDEN * DIM);
    cvt_f32_bf16<<<(HIDDEN * DIM + T - 1) / T,  T, 0, stream>>>(w2_w,   w2_bf,     HIDDEN * DIM);
    cvt_f32_bf16<<<(DIM * HIDDEN + T - 1) / T,  T, 0, stream>>>(w3_w,   w3_bf,     DIM * HIDDEN);
  }

  const size_t shmem_norm = (size_t)BALL * DIM * 2 + 256 * sizeof(float) + 4 * sizeof(float);
  (void)hipFuncSetAttribute((const void*)norm_pe_qkv_kernel,
                            hipFuncAttributeMaxDynamicSharedMemorySize, (int)shmem_norm);
  (void)hipFuncSetAttribute((const void*)mlp_gate_kernel,
                            hipFuncAttributeMaxDynamicSharedMemorySize, (int)shmem_norm);

  // 2) norm + pe + QKV (head-major split outputs)
  norm_pe_qkv_kernel<<<nballs, 256, shmem_norm, stream>>>(
      x, pos, norm1_w, pe_w, pe_b, qkv_b, qkv_w_bf, Qh, Kh, Vth, N, nballs);
  // 3) ball attention (one block per ball*head), TDM-staged tiles
  ball_attn_kernel<<<nballs * HEADS, 256, 0, stream>>>(
      Qh, Kh, Vth, pos, sigma, attn_act, N, nballs);
  // 4) proj + residual
  proj_res_kernel<<<nballs, 256, 0, stream>>>(attn_act, proj_w_bf, proj_b, x, x1);
  // 5) SwiGLU gate
  mlp_gate_kernel<<<nballs, 256, shmem_norm, stream>>>(
      x1, norm2_w, w1_bf, w1_b, w2_bf, w2_b, g_act);
  // 6) w3 + residual -> out
  mlp_out_kernel<<<nballs, 256, 0, stream>>>(g_act, w3_bf, w3_b, x1, (float*)d_out);
}
